// convAttention_34746285424731
// MI455X (gfx1250) — matrix operand order repaired and checked by reading, not compiled, not run on hardware
//
#include <hip/hip_runtime.h>

// ---------------- problem constants ----------------
#define HWSZ   3136          // 56*56
#define WIDTH  56
#define CDIM   192
#define BATCH  16
#define MTOT   (BATCH*HWSZ)  // 50176 pixels total
#define HEADS  6
#define DH     32
#define WIN    7
#define PADR   3
#define NNB    49            // WIN*WIN

typedef __attribute__((ext_vector_type(16))) __bf16 v16bf;
typedef __attribute__((ext_vector_type(8)))  __bf16 v8bf;
typedef __attribute__((ext_vector_type(8)))  float  v8f;

union FragBF { v16bf v; v8bf h[2]; };

static __device__ __forceinline__ unsigned pack_bf2(float a, float b) {
    union { __bf16 h[2]; unsigned u; } t;
    t.h[0] = (__bf16)a; t.h[1] = (__bf16)b;
    return t.u;
}

// Async copy of 64 contiguous bytes global -> LDS (4x b128).
// ISA 15.18.3: INST_OFFSET is added to BOTH the LDS and global addresses,
// so one base pair + offsets covers the run. Tracked by ASYNCcnt.
static __device__ __forceinline__ void async_tile64(unsigned lds_addr,
                                                    const void* gptr) {
    asm volatile(
        "global_load_async_to_lds_b128 %0, %1, off\n\t"
        "global_load_async_to_lds_b128 %0, %1, off offset:16\n\t"
        "global_load_async_to_lds_b128 %0, %1, off offset:32\n\t"
        "global_load_async_to_lds_b128 %0, %1, off offset:48"
        :: "v"(lds_addr), "v"(gptr)
        : "memory");
}
static __device__ __forceinline__ void wait_async0() {
    asm volatile("s_wait_asynccnt 0x0" ::: "memory");
}

// =====================================================================
// GEMM: D[n, m] = sum_k W[n,k] * X[k,m]  (pixels m are the WMMA N dim)
// MODE 0 (QKV): X = x fp32 [b][c][s] (m = b*HW+s); out bf16 [n][m],
//               epilogue adds qkv_b and scales q rows (n<192) by dh^-0.5
// MODE 1 (proj): X = attn bf16 PIXEL-major [m][192]; X tile staged with
//               global_load_async_to_lds_b128; out fp32 d_out + proj_b
// Block: 256 thr = 8 waves, tile 32n x 256m; wave tile 16n x 64m (4 WMMA).
// K loop: 6 steps of 32 (K=192), ping-pong LDS, pipelined global fetch.
// =====================================================================
template<int MODE>
__global__ __launch_bounds__(256) void gemm_wmma(
    const float*  __restrict__ Wt,    // [NOUT][192] row-major
    const float*  __restrict__ Xf,    // MODE 0 source
    const __bf16* __restrict__ Xb,    // MODE 1 source [m][192]
    const float*  __restrict__ bias,  // [NOUT]
    __bf16*       __restrict__ outb,  // MODE 0 dest
    float*        __restrict__ outf)  // MODE 1 dest
{
    // K padded to 40 elems (80B rows): 16B-aligned fragment chunks,
    // bank stride 20 dwords -> at most 2-way conflicts.
    __shared__ __align__(16) __bf16 lw[2][32][40];
    __shared__ __align__(16) __bf16 lx[2][256][40];

    const int tid   = threadIdx.x;
    const int lane  = tid & 31;
    const int wave  = tid >> 5;
    const int wn    = wave >> 2;    // 0..1 : n sub-tile
    const int wm    = wave & 3;     // 0..3 : m sub-tile (64 px each)
    const int khalf = lane >> 4;
    const int l15   = lane & 15;

    const int m0 = blockIdx.x * 256;
    const int n0 = blockIdx.y * 32;

    // ---- hoisted per-thread staging addresses ----
    // X staging: this thread owns pixel column m0+tid (k-slice per step).
    const int mcol = m0 + tid;
    const float*  xcol = nullptr;   // MODE 0: + k*HWSZ per element
    const __bf16* bcol = nullptr;   // MODE 1: + k per element (row-major)
    if (MODE == 0) {
        int bm = mcol / HWSZ, sm = mcol - bm * HWSZ;
        xcol = Xf + (size_t)(bm * CDIM) * HWSZ + sm;
    } else {
        bcol = Xb + (size_t)mcol * CDIM;
    }
    // raw LDS byte offsets of this thread's staging rows (low 32 bits of
    // the generic pointer = LDS address per the aperture scheme)
    unsigned lxoff0 = (unsigned)(unsigned long long)&lx[0][tid][0];
    unsigned lxoff1 = (unsigned)(unsigned long long)&lx[1][tid][0];

    // W staging: 2 pairs per thread
    const int wp0 = tid, wp1 = tid + 256;
    const float* wrow0 = Wt + (n0 + (wp0 >> 4)) * CDIM + 2 * (wp0 & 15);
    const float* wrow1 = Wt + (n0 + (wp1 >> 4)) * CDIM + 2 * (wp1 & 15);

    float xreg[16][2];
    float wreg[2][2];

    v8f acc[4] = {};

#define FETCH_W(KT)                                                        \
    do {                                                                   \
        const int k0f = (KT) * 32;                                         \
        wreg[0][0] = wrow0[k0f]; wreg[0][1] = wrow0[k0f + 1];              \
        wreg[1][0] = wrow1[k0f]; wreg[1][1] = wrow1[k0f + 1];              \
    } while (0)
#define COMMIT_W(BUF)                                                      \
    do {                                                                   \
        *(unsigned*)&lw[BUF][wp0 >> 4][2 * (wp0 & 15)] =                   \
            pack_bf2(wreg[0][0], wreg[0][1]);                              \
        *(unsigned*)&lw[BUF][wp1 >> 4][2 * (wp1 & 15)] =                   \
            pack_bf2(wreg[1][0], wreg[1][1]);                              \
    } while (0)
#define FETCH_X(KT)                                                        \
    do {                                                                   \
        const int k0f = (KT) * 32;                                         \
        _Pragma("unroll")                                                  \
        for (int i = 0; i < 16; ++i) {                                     \
            const int k = k0f + 2 * i;                                     \
            xreg[i][0] = xcol[(size_t)k * HWSZ];                           \
            xreg[i][1] = xcol[(size_t)(k + 1) * HWSZ];                     \
        }                                                                  \
    } while (0)
#define COMMIT_X(BUF)                                                      \
    do {                                                                   \
        _Pragma("unroll")                                                  \
        for (int i = 0; i < 16; ++i)                                       \
            *(unsigned*)&lx[BUF][tid][2 * i] =                             \
                pack_bf2(xreg[i][0], xreg[i][1]);                          \
    } while (0)

    // ---- prologue: stage K-step 0 into buffer 0 ----
    FETCH_W(0);
    if (MODE == 0) { FETCH_X(0); }
    COMMIT_W(0);
    if (MODE == 0) {
        COMMIT_X(0);
    } else {
        async_tile64(lxoff0, bcol);
        wait_async0();
    }
    __syncthreads();

    #pragma unroll
    for (int kt = 0; kt < 6; ++kt) {
        const int buf = kt & 1;
        if (kt < 5) {                      // next tile's loads in flight
            FETCH_W(kt + 1);
            if (MODE == 0) {
                FETCH_X(kt + 1);
            } else {
                async_tile64(buf ? lxoff0 : lxoff1, bcol + (kt + 1) * 32);
            }
        }

        // A fragment: lane M=l15, K = khalf*16 + 0..15
        FragBF a;
        a.h[0] = *(const v8bf*)&lw[buf][wn * 16 + l15][khalf * 16];
        a.h[1] = *(const v8bf*)&lw[buf][wn * 16 + l15][khalf * 16 + 8];
        #pragma unroll
        for (int s = 0; s < 4; ++s) {
            // B fragment: lane N=l15 (pixel), K = khalf*16 + 0..15
            FragBF bb;
            const int mr = wm * 64 + s * 16 + l15;
            bb.h[0] = *(const v8bf*)&lx[buf][mr][khalf * 16];
            bb.h[1] = *(const v8bf*)&lx[buf][mr][khalf * 16 + 8];
            acc[s] = __builtin_amdgcn_wmma_f32_16x16x32_bf16(
                         false, a.v, false, bb.v, (short)0, acc[s],
                         false, false);
        }

        if (kt < 5) {
            COMMIT_W(buf ^ 1);
            if (MODE == 0) { COMMIT_X(buf ^ 1); }
        }
        if (MODE == 1) { wait_async0(); }  // async writes landed in LDS
        __syncthreads();
    }
#undef FETCH_W
#undef COMMIT_W
#undef FETCH_X
#undef COMMIT_X

    // ---- epilogue: D lane holds N=l15, M = r + 8*khalf ----
    const bool qscale = (MODE == 0) && (n0 < CDIM);   // block-uniform
    #pragma unroll
    for (int r = 0; r < 8; ++r) {
        const int ng = n0 + wn * 16 + r + 8 * khalf;
        const float bv = bias[ng];
        #pragma unroll
        for (int s = 0; s < 4; ++s) {
            float v = acc[s][r] + bv;
            const int mg = m0 + wm * 64 + s * 16 + l15;
            if (MODE == 0) {
                if (qscale) v *= 0.17677669529663687f;
                outb[(size_t)ng * MTOT + mg] = (__bf16)v;
            } else {
                const int bi = mg / HWSZ, si = mg - bi * HWSZ;
                outf[(bi * CDIM + ng) * HWSZ + si] = v;
            }
        }
    }
}

// =====================================================================
// Neighborhood attention: 14x14 pixel tile / block, 20x20 K/V halo in LDS,
// one pixel per thread, online softmax over 49 neighbors.
// Output is PIXEL-major [m][192] so each thread stores one contiguous
// 64B run (and the proj GEMM can stage it with async b128 copies).
// =====================================================================
#define TILE 14
#define HALO 20
#define DPAD 40   // 80B rows -> 16B-aligned v8bf chunks, 2-way banks max

__global__ __launch_bounds__(256) void neigh_attn(
    const __bf16* __restrict__ qkvt,        // [576][MTOT]
    const float*  __restrict__ bias_table,  // [49][HEADS]
    __bf16*       __restrict__ attnt)       // [MTOT][192] pixel-major
{
    __shared__ __align__(16) __bf16 lk[HALO][HALO][DPAD];
    __shared__ __align__(16) __bf16 lv[HALO][HALO][DPAD];
    __shared__ float lbias[NNB];

    const int tid = threadIdx.x;
    const int tx = blockIdx.x & 3, ty = blockIdx.x >> 2;
    const int h  = blockIdx.y;
    const int b  = blockIdx.z;
    const int x0 = tx * TILE, y0 = ty * TILE;

    if (tid < NNB) lbias[tid] = bias_table[tid * HEADS + h];

    const size_t kbase = (size_t)(CDIM     + h * DH) * MTOT + (size_t)b * HWSZ;
    const size_t vbase = (size_t)(2 * CDIM + h * DH) * MTOT + (size_t)b * HWSZ;
    for (int idx = tid; idx < HALO * HALO * DH; idx += 256) {
        int hx = idx % HALO;
        int t2 = idx / HALO;
        int hy = t2 % HALO;
        int d  = t2 / HALO;
        int gy = y0 + hy - PADR, gx = x0 + hx - PADR;
        bool ok = (gy >= 0) && (gy < WIDTH) && (gx >= 0) && (gx < WIDTH);
        int gs = gy * WIDTH + gx;
        lk[hy][hx][d] = ok ? qkvt[kbase + (size_t)d * MTOT + gs] : (__bf16)0.f;
        lv[hy][hx][d] = ok ? qkvt[vbase + (size_t)d * MTOT + gs] : (__bf16)0.f;
    }
    __syncthreads();

    if (tid >= TILE * TILE) return;
    const int px = tid % TILE, py = tid / TILE;
    const int gs = (y0 + py) * WIDTH + (x0 + px);

    float q[DH];
    #pragma unroll
    for (int d = 0; d < DH; ++d)   // q pre-scaled in QKV epilogue
        q[d] = (float)qkvt[(size_t)(h * DH + d) * MTOT + (size_t)b * HWSZ + gs];

    float acc[DH];
    #pragma unroll
    for (int d = 0; d < DH; ++d) acc[d] = 0.f;
    float mv = -3.0e38f, ssum = 0.f;

    for (int i = 0; i < NNB; ++i) {
        int dy = i / WIN, dx = i - dy * WIN;
        const v8bf* kc = (const v8bf*)&lk[py + dy][px + dx][0];
        const v8bf* vc = (const v8bf*)&lv[py + dy][px + dx][0];
        float s = lbias[i];
        #pragma unroll
        for (int c = 0; c < 4; ++c) {
            v8bf kk = kc[c];
            #pragma unroll
            for (int j = 0; j < 8; ++j) s += q[c * 8 + j] * (float)kk[j];
        }
        float mn   = fmaxf(mv, s);
        float corr = __expf(mv - mn);   // first iter: exp(-inf) = 0
        float w    = __expf(s - mn);
        ssum = ssum * corr + w;
        #pragma unroll
        for (int c = 0; c < 4; ++c) {
            v8bf vv = vc[c];
            #pragma unroll
            for (int j = 0; j < 8; ++j)
                acc[c * 8 + j] = acc[c * 8 + j] * corr + w * (float)vv[j];
        }
        mv = mn;
    }
    float inv = 1.0f / ssum;
    // contiguous 64B store run: [m][h*32 .. h*32+31]
    const size_t gm = (size_t)b * HWSZ + gs;
    #pragma unroll
    for (int d = 0; d < DH; ++d)
        attnt[gm * CDIM + h * DH + d] = (__bf16)(acc[d] * inv);
}

// =====================================================================
extern "C" void kernel_launch(void* const* d_in, const int* in_sizes, int n_in,
                              void* d_out, int out_size, void* d_ws, size_t ws_size,
                              hipStream_t stream) {
    const float* x          = (const float*)d_in[0];
    const float* qkv_w      = (const float*)d_in[1];
    const float* qkv_b      = (const float*)d_in[2];
    const float* proj_w     = (const float*)d_in[3];
    const float* proj_b     = (const float*)d_in[4];
    const float* bias_table = (const float*)d_in[5];
    float* out = (float*)d_out;

    // workspace: qkv bf16 [576][50176] then attn bf16 [50176][192] (~77 MB)
    __bf16* qkvt  = (__bf16*)d_ws;
    __bf16* attnt = qkvt + (size_t)3 * CDIM * MTOT;

    // 1) QKV projection GEMM (WMMA bf16), q pre-scaled, bf16 out
    gemm_wmma<0><<<dim3(MTOT / 256, (3 * CDIM) / 32), 256, 0, stream>>>(
        qkv_w, x, nullptr, qkv_b, qkvt, nullptr);

    // 2) neighborhood attention (online softmax), bf16 pixel-major out
    neigh_attn<<<dim3(16, HEADS, BATCH), 256, 0, stream>>>(
        qkvt, bias_table, attnt);

    // 3) output projection GEMM (WMMA bf16, async-LDS staging), fp32 out
    gemm_wmma<1><<<dim3(MTOT / 256, CDIM / 32), 256, 0, stream>>>(
        proj_w, nullptr, attnt, proj_b, nullptr, out);
}
